// TransConvLayer_60610578481376
// MI455X (gfx1250) — matrix operand order repaired and checked by reading, not compiled, not run on hardware
//
#include <hip/hip_runtime.h>

typedef __attribute__((ext_vector_type(16))) __bf16 v16bf;
typedef __attribute__((ext_vector_type(8)))  __bf16 v8bf;
typedef __attribute__((ext_vector_type(4)))  __bf16 v4bf;
typedef __attribute__((ext_vector_type(8)))  float  v8f;

#define N_ROWS  262144
#define EPS_F   1e-8f

// workspace byte offsets
#define WS_RED   0                      // float[8] : [0]=sum q^2, [1]=sum k^2
#define WS_KSUM  64                     // float[256] : ks_sum_raw[h*64+m]
#define WS_KVS   2048                   // float[4*64*64] : kvs_raw
#define WS_WB    131072                 // __bf16[3*256*256] weights (q,k,v)
#define WS_QWS   524288                 // __bf16[N*256] raw q (+bias)
#define WS_VWS   (524288 + 134217728)   // __bf16[N*256] raw v (+bias)

static __device__ __forceinline__ v16bf cat8(v8bf a, v8bf b) {
  return __builtin_shufflevector(a, b, 0,1,2,3,4,5,6,7,8,9,10,11,12,13,14,15);
}

// ---------------- kernel 0: zero reductions + convert weights to bf16 ----------------
__global__ void __launch_bounds__(256)
k_init(const float* __restrict__ Wq, const float* __restrict__ Wk,
       const float* __restrict__ Wv, char* __restrict__ ws) {
  int tid = blockIdx.x * 256 + threadIdx.x;          // 0 .. 196607
  float* zf = (float*)ws;
  if (tid < 16896) zf[tid] = 0.0f;                   // covers RED, KSUM, KVS regions
  const float* src = (tid < 65536) ? Wq : (tid < 131072 ? Wk : Wv);
  __bf16* Wb = (__bf16*)(ws + WS_WB);
  Wb[tid] = (__bf16)src[tid & 65535];
}

// ---------------- kernel 1: projections + global reductions ----------------
__global__ void __launch_bounds__(256)
k_passA(const float* __restrict__ Xq, const float* __restrict__ Xs,
        const float* __restrict__ bq, const float* __restrict__ bk,
        const float* __restrict__ bvp, char* __restrict__ ws) {
  __shared__ __bf16 sXq[64][256];     // 32KB  staged query rows (bf16)
  __shared__ __bf16 sXs[64][256];     // 32KB  staged source rows
  __shared__ __bf16 skT[256][64];     // 32KB  k transposed  [m][n]
  __shared__ __bf16 svT[256][64];     // 32KB  v transposed  [d][n]
  __shared__ float  sKsum[256];
  __shared__ float  sBias[768];

  float*        red   = (float*)(ws + WS_RED);
  float*        ksumG = (float*)(ws + WS_KSUM);
  float*        kvsG  = (float*)(ws + WS_KVS);
  const __bf16* Wb    = (const __bf16*)(ws + WS_WB);
  __bf16*       qws   = (__bf16*)(ws + WS_QWS);
  __bf16*       vws   = (__bf16*)(ws + WS_VWS);

  const int t = threadIdx.x;
  const int w = t >> 5, lane = t & 31;
  const int ll = lane & 15, lh = lane >> 4;

  if (t < 256) { sBias[t] = bq[t]; sBias[256 + t] = bk[t]; sBias[512 + t] = bvp[t]; sKsum[t] = 0.0f; }
  __syncthreads();

  // persistent kvs accumulators: wave handles pairs pi = w*2 + p : h = pi>>2, mtile = pi&3
  v8f kvsA[2][4];
  for (int p = 0; p < 2; ++p)
    for (int dt = 0; dt < 4; ++dt)
      for (int j = 0; j < 8; ++j) kvsA[p][dt][j] = 0.0f;

  float ssq_q = 0.0f, ssq_k = 0.0f;

  for (int slab = blockIdx.x; slab < (N_ROWS / 64); slab += gridDim.x) {
    const int rowbase = slab * 64;

    // ---- stage 64 rows of Xq and Xs as bf16 into LDS ----
#pragma unroll
    for (int i = 0; i < 16; ++i) {
      int f4  = i * 256 + t;
      int row = f4 >> 6, c4 = (f4 & 63) << 2;
      float4 a = *(const float4*)(Xq + (size_t)(rowbase + row) * 256 + c4);
      float4 b = *(const float4*)(Xs + (size_t)(rowbase + row) * 256 + c4);
      v4bf av = { (__bf16)a.x, (__bf16)a.y, (__bf16)a.z, (__bf16)a.w };
      v4bf bv = { (__bf16)b.x, (__bf16)b.y, (__bf16)b.z, (__bf16)b.w };
      *(v4bf*)&sXq[row][c4] = av;
      *(v4bf*)&sXs[row][c4] = bv;
    }
    __syncthreads();

    // ---- GEMM: wave owns 6 column tiles (c in [w*6, w*6+6)) over q|k|v ----
    for (int c6 = 0; c6 < 6; ++c6) {
      const int c   = w * 6 + c6;
      const int mat = c >> 4;             // 0=q 1=k 2=v
      const int oc0 = (c & 15) << 4;
      const __bf16* wb = Wb + (size_t)mat * 65536 + (size_t)(oc0 + ll) * 256;
      const __bf16 (*sX)[256] = (mat == 0) ? sXq : sXs;

      v8f acc[4];
      for (int rt = 0; rt < 4; ++rt)
        for (int j = 0; j < 8; ++j) acc[rt][j] = 0.0f;

#pragma unroll
      for (int kc = 0; kc < 8; ++kc) {
        // B fragment: lane holds col oc0+ll, K = kc*32 + lh*8 + (0..7) and +16
        v16bf bfrag = cat8(*(const v8bf*)(wb + kc * 32 + lh * 8), *(const v8bf*)(wb + kc * 32 + lh * 8 + 16));
#pragma unroll
        for (int rt = 0; rt < 4; ++rt) {
          const __bf16* xr = &sX[rt * 16 + ll][kc * 32 + lh * 8];
          v8bf a0 = *(const v8bf*)xr;          // K = kc*32 + lh*8 + (0..7)
          v8bf a1 = *(const v8bf*)(xr + 16);   // K = +16
          acc[rt] = __builtin_amdgcn_wmma_f32_16x16x32_bf16(
              false, cat8(a0, a1), false, bfrag, (short)0, acc[rt], false, false);
        }
      }

      const float bias = sBias[mat * 256 + oc0 + ll];
#pragma unroll
      for (int rt = 0; rt < 4; ++rt) {
        float bv8[8];
#pragma unroll
        for (int j = 0; j < 8; ++j) bv8[j] = acc[rt][j] + bias;
        if (mat == 0) {            // q: sumsq + store raw bf16
          float s = 0.0f;
          for (int j = 0; j < 8; ++j) s += bv8[j] * bv8[j];
          ssq_q += s;
          for (int j = 0; j < 8; ++j)
            qws[(size_t)(rowbase + rt * 16 + j + 8 * lh) * 256 + oc0 + ll] = (__bf16)bv8[j];
        } else if (mat == 1) {     // k: sumsq + column sums + kT to LDS
          float s = 0.0f, cs = 0.0f;
          for (int j = 0; j < 8; ++j) { s += bv8[j] * bv8[j]; cs += bv8[j]; }
          ssq_k += s;
          atomicAdd(&sKsum[oc0 + ll], cs);
          v8bf pk;
          for (int j = 0; j < 8; ++j) pk[j] = (__bf16)bv8[j];
          *(v8bf*)&skT[oc0 + ll][rt * 16 + 8 * lh] = pk;
        } else {                   // v: store raw bf16 + vT to LDS
          for (int j = 0; j < 8; ++j)
            vws[(size_t)(rowbase + rt * 16 + j + 8 * lh) * 256 + oc0 + ll] = (__bf16)bv8[j];
          v8bf pk;
          for (int j = 0; j < 8; ++j) pk[j] = (__bf16)bv8[j];
          *(v8bf*)&svT[oc0 + ll][rt * 16 + 8 * lh] = pk;
        }
      }
    }
    __syncthreads();

    // ---- kvs[h][m][d] += k^T v over this slab's 64 rows, via WMMA ----
#pragma unroll
    for (int p = 0; p < 2; ++p) {
      const int pi = w * 2 + p;
      const int h = pi >> 2, mt = pi & 3;
      const __bf16* krow = &skT[h * 64 + mt * 16 + ll][0];   // A row m, K = n
#pragma unroll
      for (int kc = 0; kc < 2; ++kc) {
        v8bf a0 = *(const v8bf*)(krow + kc * 32 + lh * 8);
        v8bf a1 = *(const v8bf*)(krow + kc * 32 + 16 + lh * 8);
        v16bf afrag = cat8(a0, a1);
#pragma unroll
        for (int dt = 0; dt < 4; ++dt) {
          v16bf bfrag = cat8(*(const v8bf*)&svT[h * 64 + dt * 16 + ll][kc * 32 + lh * 8], *(const v8bf*)&svT[h * 64 + dt * 16 + ll][kc * 32 + lh * 8 + 16]);
          kvsA[p][dt] = __builtin_amdgcn_wmma_f32_16x16x32_bf16(
              false, afrag, false, bfrag, (short)0, kvsA[p][dt], false, false);
        }
      }
    }
    __syncthreads();
  }

  // ---- block epilogue: push reductions ----
#pragma unroll
  for (int p = 0; p < 2; ++p) {
    const int pi = w * 2 + p;
    const int h = pi >> 2, mt = pi & 3;
    for (int dt = 0; dt < 4; ++dt)
      for (int j = 0; j < 8; ++j)
        atomicAdd(&kvsG[h * 4096 + (mt * 16 + j + 8 * lh) * 64 + dt * 16 + ll],
                  kvsA[p][dt][j]);
  }
  float sq = ssq_q, sk = ssq_k;
  for (int off = 16; off > 0; off >>= 1) {
    sq += __shfl_xor(sq, off, 32);
    sk += __shfl_xor(sk, off, 32);
  }
  if (lane == 0) { atomicAdd(&red[0], sq); atomicAdd(&red[1], sk); }
  __syncthreads();
  if (t < 256) atomicAdd(&ksumG[t], sKsum[t]);
}

// ---------------- kernel 2: per-row attention combine ----------------
__global__ void __launch_bounds__(256)
k_passB(char* __restrict__ ws, float* __restrict__ out) {
  __shared__ __bf16 sKvsT[4][64][64];   // [h][d][m] bf16, 32KB
  __shared__ float  sKsumB[256];
  __shared__ float  sT[8][16][64];      // per-wave q@kvs tile, 32KB

  const float*  red   = (const float*)(ws + WS_RED);
  const float*  kvsG  = (const float*)(ws + WS_KVS);
  const float*  ksumG = (const float*)(ws + WS_KSUM);
  const __bf16* qws   = (const __bf16*)(ws + WS_QWS);
  const __bf16* vws   = (const __bf16*)(ws + WS_VWS);

  const int t = threadIdx.x, w = t >> 5, lane = t & 31;
  const int ll = lane & 15, lh = lane >> 4;

  for (int i = 0; i < 64; ++i) {
    int idx = i * 256 + t;              // 0..16383
    int h = idx >> 12, rem = idx & 4095;
    int m = rem >> 6, d = rem & 63;
    sKvsT[h][d][m] = (__bf16)kvsG[idx];
  }
  if (t < 256) sKsumB[t] = ksumG[t];
  __syncthreads();

  const float sqk = 1.0f / ((sqrtf(red[0]) + EPS_F) * (sqrtf(red[1]) + EPS_F));
  const float Nf = 262144.0f;

  const int r0 = blockIdx.x * 128 + w * 16;
  float accv[32];
  for (int u = 0; u < 32; ++u) accv[u] = 0.0f;

  for (int h = 0; h < 4; ++h) {
    // A fragments from raw q + normalizer dot from the same data
    v16bf afr[2];
    float dp = 0.0f;
#pragma unroll
    for (int kc = 0; kc < 2; ++kc) {
      const __bf16* qp = qws + (size_t)(r0 + ll) * 256 + h * 64 + kc * 32 + lh * 8;
      v8bf a0 = *(const v8bf*)qp;
      v8bf a1 = *(const v8bf*)(qp + 16);
      afr[kc] = cat8(a0, a1);
#pragma unroll
      for (int j = 0; j < 8; ++j) {
        dp += (float)a0[j] * sKsumB[h * 64 + kc * 32 + lh * 8 + j];
        dp += (float)a1[j] * sKsumB[h * 64 + kc * 32 + 16 + lh * 8 + j];
      }
    }
    dp += __shfl_xor(dp, 16, 32);       // full 64-dot for row r0+ll

    // q_raw @ kvs_raw  (scale folded in later)
#pragma unroll
    for (int dt = 0; dt < 4; ++dt) {
      v8f cacc;
      for (int j = 0; j < 8; ++j) cacc[j] = 0.0f;
#pragma unroll
      for (int kc = 0; kc < 2; ++kc) {
        v16bf bfr = cat8(*(const v8bf*)&sKvsT[h][dt * 16 + ll][kc * 32 + lh * 8], *(const v8bf*)&sKvsT[h][dt * 16 + ll][kc * 32 + lh * 8 + 16]);
        cacc = __builtin_amdgcn_wmma_f32_16x16x32_bf16(
            false, afr[kc], false, bfr, (short)0, cacc, false, false);
      }
      for (int j = 0; j < 8; ++j) sT[w][j + 8 * lh][dt * 16 + ll] = cacc[j];
    }
    asm volatile("s_wait_dscnt 0x0" ::: "memory");  // wave-local LDS transpose handoff

    const float rcp = 1.0f / (sqk * dp + Nf);
    const __bf16* vp = vws + (size_t)(r0 + ll) * 256 + h * 64 + lh * 32;
    v16bf vv0 = *(const v16bf*)vp;
    v16bf vv1 = *(const v16bf*)(vp + 16);
#pragma unroll
    for (int u = 0; u < 16; ++u)
      accv[u] += (sqk * sT[w][ll][lh * 32 + u] + Nf * (float)vv0[u]) * rcp;
#pragma unroll
    for (int u = 0; u < 16; ++u)
      accv[16 + u] += (sqk * sT[w][ll][lh * 32 + 16 + u] + Nf * (float)vv1[u]) * rcp;
  }

  float* op = out + (size_t)(r0 + ll) * 64 + lh * 32;
#pragma unroll
  for (int u = 0; u < 32; u += 4) {
    float4 o = { accv[u] * 0.25f, accv[u + 1] * 0.25f,
                 accv[u + 2] * 0.25f, accv[u + 3] * 0.25f };
    *(float4*)(op + u) = o;
  }
}

extern "C" void kernel_launch(void* const* d_in, const int* in_sizes, int n_in,
                              void* d_out, int out_size, void* d_ws, size_t ws_size,
                              hipStream_t stream) {
  (void)in_sizes; (void)n_in; (void)out_size; (void)ws_size;
  const float* Xq = (const float*)d_in[0];
  const float* Xs = (const float*)d_in[1];
  const float* Wq = (const float*)d_in[2];
  const float* bq = (const float*)d_in[3];
  const float* Wk = (const float*)d_in[4];
  const float* bk = (const float*)d_in[5];
  const float* Wv = (const float*)d_in[6];
  const float* bv = (const float*)d_in[7];
  char* ws = (char*)d_ws;

  k_init <<<768, 256, 0, stream>>>(Wq, Wk, Wv, ws);
  k_passA<<<1024, 256, 0, stream>>>(Xq, Xs, bq, bk, bv, ws);
  k_passB<<<2048, 256, 0, stream>>>(ws, (float*)d_out);
}
